// TransformerBlock_89017492177564
// MI455X (gfx1250) — matrix operand order repaired and checked by reading, not compiled, not run on hardware
//
#include <hip/hip_runtime.h>
#include <hip/hip_bf16.h>
#include <math.h>

// ---------------------------------------------------------------------------
// Problem constants (match reference)
// ---------------------------------------------------------------------------
#define BB   2
#define SS   2048
#define DD   1024
#define HH   16
#define HD   64
#define FF_  4096
#define BLK  128
#define NTOK (BB * SS)          // 4096 tokens

typedef __bf16 bf16_t;
typedef __attribute__((ext_vector_type(16))) bf16_t v16bf;
typedef __attribute__((ext_vector_type(8)))  float  v8f;
typedef __attribute__((ext_vector_type(8)))  unsigned int v8u;
typedef __attribute__((ext_vector_type(4)))  unsigned int u32x4;
typedef __attribute__((ext_vector_type(8)))  int i32x8;
typedef __attribute__((ext_vector_type(4)))  int i32x4;

union FragCast { v8u u; v16bf b; };

__device__ __forceinline__ unsigned short f2bf(float f) {
    unsigned int u = __float_as_uint(f);
    u = (u + 0x7FFFu + ((u >> 16) & 1u)) >> 16;   // round-to-nearest-even
    return (unsigned short)u;
}

__device__ __forceinline__ v8f wmma_bf16(v16bf a, v16bf b, v8f c) {
    return __builtin_amdgcn_wmma_f32_16x16x32_bf16(
        /*neg_a=*/false, a, /*neg_b=*/false, b,
        /*c_mod=*/(short)0, c, /*reuse_a=*/false, /*reuse_b=*/false);
}

// A-matrix fragment: 16x32 (MxK) bf16, row-major source with leading dim ld.
__device__ __forceinline__ v16bf load_a_frag(const unsigned short* p, int ld, int lane) {
    v8u u;
    const int row  = lane & 15;
    const int koff = (lane >> 4) << 4;
    const unsigned short* rp = p + row * ld;
#pragma unroll
    for (int v = 0; v < 8; ++v) {
        int kk = (v << 1) + koff;
        u[v] = *(const unsigned int*)(rp + kk);
    }
    FragCast c; c.u = u; return c.b;
}

// B-matrix fragment: 32x16 (KxN) bf16, from row-major [N, K] storage.
__device__ __forceinline__ v16bf load_b_frag_nk(const unsigned short* p, int ld, int lane) {
    v8u u;
    const int col  = lane & 15;
    const int koff = (lane >> 4) << 4;
    const unsigned short* rp = p + col * ld + koff;
#pragma unroll
    for (int v = 0; v < 8; ++v) {
        u[v] = *(const unsigned int*)(rp + (v << 1));
    }
    FragCast c; c.u = u; return c.b;
}

// ---------------------------------------------------------------------------
// Tensor Data Mover: 2-D bf16 tile (tileK x tileN) from row-major [N,K]
// global memory into LDS (contiguous tileN x tileK, row-major, ld = tileK).
// D# built per CDNA5 ISA 8.3/8.4: group0 = {count/type/lds/global},
// group1 = {mask,data_size,dims,tile dims,stride}.
// ---------------------------------------------------------------------------
__device__ __forceinline__ void tdm_load_tile_bf16(
    const unsigned short* gsrc,     // tile start in global memory
    unsigned ldsByteOff,            // destination LDS byte offset
    unsigned tileK, unsigned tileN, // tile dims (dim0 = K contiguous, dim1 = rows)
    unsigned tensorK, unsigned tensorN,
    unsigned strideK)               // row stride in elements
{
    unsigned long long ga = (unsigned long long)(size_t)gsrc;
    u32x4 g0;
    g0[0] = 1u;                                        // count=1, user mode
    g0[1] = ldsByteOff;                                // lds_addr
    g0[2] = (unsigned)(ga & 0xFFFFFFFFu);              // global_addr[31:0]
    g0[3] = (unsigned)((ga >> 32) & 0x1FFFFFFu) | (2u << 30);  // addr[56:32] | type=2
    i32x8 g1;
    g1[0] = (int)(1u << 16);                           // workgroup_mask=0, data_size=1 (2B)
    g1[1] = (int)((tensorK & 0xFFFFu) << 16);          // tensor_dim0[15:0]
    g1[2] = (int)(((tensorK >> 16) & 0xFFFFu) | ((tensorN & 0xFFFFu) << 16));
    g1[3] = (int)(((tensorN >> 16) & 0xFFFFu) | ((tileK & 0xFFFFu) << 16)); // tile_dim0
    g1[4] = (int)(tileN & 0xFFFFu);                    // tile_dim1 (tile_dim2 = 0)
    g1[5] = (int)strideK;                              // tensor_dim0_stride[31:0]
    g1[6] = 0;
    g1[7] = 0;
    i32x4 z4 = {0, 0, 0, 0};
    i32x8 z8 = {0, 0, 0, 0, 0, 0, 0, 0};
    __builtin_amdgcn_tensor_load_to_lds(g0, g1, z4, z4, z8, 0);
}

// ---------------------------------------------------------------------------
// fp32 -> bf16 cast (weights, once per call)
// ---------------------------------------------------------------------------
__global__ void cast_bf16_kernel(const float* __restrict__ src,
                                 unsigned short* __restrict__ dst, int n) {
    int i = blockIdx.x * blockDim.x + threadIdx.x;
    if (i < n) dst[i] = f2bf(src[i]);
}

// ---------------------------------------------------------------------------
// LayerNorm -> bf16
// ---------------------------------------------------------------------------
__global__ void ln_bf16_kernel(const float* __restrict__ x,
                               const float* __restrict__ g,
                               const float* __restrict__ b,
                               unsigned short* __restrict__ out, int D) {
    const int row = blockIdx.x;
    const int tid = threadIdx.x;
    __shared__ float s1[256], s2[256];
    float a = 0.f, sq = 0.f;
    for (int i = tid; i < D; i += 256) {
        float v = x[row * D + i];
        a += v; sq += v * v;
    }
    s1[tid] = a; s2[tid] = sq;
    __syncthreads();
    for (int off = 128; off > 0; off >>= 1) {
        if (tid < off) { s1[tid] += s1[tid + off]; s2[tid] += s2[tid + off]; }
        __syncthreads();
    }
    const float mu   = s1[0] / (float)D;
    const float var  = s2[0] / (float)D - mu * mu;
    const float rstd = rsqrtf(var + 1e-5f);
    for (int i = tid; i < D; i += 256) {
        float v = (x[row * D + i] - mu) * rstd * g[i] + b[i];
        out[row * D + i] = f2bf(v);
    }
}

__global__ void zero_f32_kernel(float* __restrict__ p, int n) {
    int i = blockIdx.x * blockDim.x + threadIdx.x;
    if (i < n) p[i] = 0.f;
}

// ---------------------------------------------------------------------------
// WMMA GEMM with TDM double-buffered tiles.
// C[M,N] = epi( A[M,K](bf16) * W[N,K]^T(bf16) + bias )
// block = 256 threads (8 waves); tile 128x64; wave -> 32x32 (2x2 WMMAs).
// EPI: 0 = QKV scatter, 1 = exact GELU -> bf16, 2 = +resid -> fp32
// ---------------------------------------------------------------------------
struct GemmArgs {
    const unsigned short* A;   // [M,K] bf16
    const unsigned short* W;   // [N,K] bf16
    const float* bias;         // [N]
    const float* resid;        // [M,N] fp32 (EPI 2)
    float* outf;               // [M,N] fp32 (EPI 2)
    unsigned short* outb;      // [M,N] bf16 (EPI 1)
    unsigned short* q;         // (EPI 0)
    unsigned short* k;
    unsigned short* v;
    int M, N, K;
};

#define ATILE (128 * 32)   // elements per A buffer
#define WTILE (64 * 32)    // elements per W buffer

template <int EPI>
__global__ __launch_bounds__(256)
void gemm_wmma_kernel(GemmArgs ga) {
    const int tid  = threadIdx.x;
    const int lane = tid & 31;
    const int wave = tid >> 5;
    const int wr   = wave >> 1;           // 0..3 : 32-row group
    const int wc   = wave & 1;            // 0..1 : 32-col group
    const int n0 = blockIdx.x * 64;
    const int mblk = blockIdx.y * 128;

    // [ Abuf0 | Abuf1 | Wbuf0 | Wbuf1 ]  (assumes smem at LDS offset 0)
    __shared__ unsigned short smem[2 * ATILE + 2 * WTILE];

    v8f acc[2][2];
#pragma unroll
    for (int i = 0; i < 2; ++i)
#pragma unroll
        for (int j = 0; j < 2; ++j) acc[i][j] = (v8f)0.f;

    const int ktiles = ga.K >> 5;

    // preload tile 0 (wave 0 issues the DMA; TDM ignores EXEC)
    if (tid < 32) {
        tdm_load_tile_bf16(ga.A + (size_t)mblk * ga.K, 0u,
                           32u, 128u, (unsigned)ga.K, (unsigned)ga.M, (unsigned)ga.K);
        tdm_load_tile_bf16(ga.W + (size_t)n0 * ga.K, (unsigned)(2 * ATILE) * 2u,
                           32u, 64u, (unsigned)ga.K, (unsigned)ga.N, (unsigned)ga.K);
    }

    for (int kt = 0; kt < ktiles; ++kt) {
        const int cur = kt & 1;
        if (tid < 32) __builtin_amdgcn_s_wait_tensorcnt(0);
        __syncthreads();   // tile `cur` visible; everyone done with buf `1-cur`

        if (kt + 1 < ktiles && tid < 32) {
            const int nxt = 1 - cur;
            tdm_load_tile_bf16(ga.A + (size_t)mblk * ga.K + (kt + 1) * 32,
                               (unsigned)(nxt * ATILE) * 2u,
                               32u, 128u, (unsigned)ga.K, (unsigned)ga.M, (unsigned)ga.K);
            tdm_load_tile_bf16(ga.W + (size_t)n0 * ga.K + (kt + 1) * 32,
                               (unsigned)(2 * ATILE + nxt * WTILE) * 2u,
                               32u, 64u, (unsigned)ga.K, (unsigned)ga.N, (unsigned)ga.K);
        }

        const unsigned short* As = smem + cur * ATILE;
        const unsigned short* Ws = smem + 2 * ATILE + cur * WTILE;

        v16bf a0 = load_a_frag(As + (wr * 32 + 0)  * 32, 32, lane);
        v16bf a1 = load_a_frag(As + (wr * 32 + 16) * 32, 32, lane);
        v16bf b0 = load_b_frag_nk(Ws + (wc * 32 + 0)  * 32, 32, lane);
        v16bf b1 = load_b_frag_nk(Ws + (wc * 32 + 16) * 32, 32, lane);

        acc[0][0] = wmma_bf16(a0, b0, acc[0][0]);
        acc[0][1] = wmma_bf16(a0, b1, acc[0][1]);
        acc[1][0] = wmma_bf16(a1, b0, acc[1][0]);
        acc[1][1] = wmma_bf16(a1, b1, acc[1][1]);
    }

    // Epilogue. D layout: lanes 0-15 -> N=lane, M=r; lanes 16-31 -> M=r+8.
    const int mbase = mblk + wr * 32 + ((lane >> 4) << 3);
    const int nbase = n0 + wc * 32 + (lane & 15);
#pragma unroll
    for (int ti = 0; ti < 2; ++ti) {
#pragma unroll
        for (int tj = 0; tj < 2; ++tj) {
            const int n = nbase + tj * 16;
            const float bias = ga.bias[n];
#pragma unroll
            for (int r = 0; r < 8; ++r) {
                const int m = mbase + ti * 16 + r;
                float val = acc[ti][tj][r] + bias;
                if (EPI == 0) {
                    int which = n >> 10;
                    int c     = n & (DD - 1);
                    int h     = c >> 6;
                    int hd    = c & 63;
                    int b_    = m >> 11;
                    int s_    = m & (SS - 1);
                    size_t di = ((size_t)(b_ * HH + h) * SS + s_) * HD + hd;
                    unsigned short bv = f2bf(val);
                    if (which == 0)      ga.q[di] = bv;
                    else if (which == 1) ga.k[di] = bv;
                    else                 ga.v[di] = bv;
                } else if (EPI == 1) {
                    float gel = 0.5f * val * (1.0f + erff(val * 0.70710678118654752f));
                    ga.outb[(size_t)m * ga.N + n] = f2bf(gel);
                } else {
                    ga.outf[(size_t)m * ga.N + n] = ga.resid[(size_t)m * ga.N + n] + val;
                }
            }
        }
    }
}

// ---------------------------------------------------------------------------
// Banded attention: one wave per (b, h, 16-query tile); window 272 keys.
// ---------------------------------------------------------------------------
#define WKEYS 272
#define WPAD  288

__global__ __launch_bounds__(32)
void attn_wmma_kernel(const unsigned short* __restrict__ qg,
                      const unsigned short* __restrict__ kg,
                      const unsigned short* __restrict__ vg,
                      unsigned short* __restrict__ ctx,
                      float* __restrict__ attnW) {
    const int lane = threadIdx.x & 31;
    const int q0   = blockIdx.x * 16;
    const int bh   = blockIdx.y;
    const int b_   = bh >> 4;
    const int h_   = bh & 15;

    const unsigned short* qh = qg + (size_t)bh * SS * HD;
    const unsigned short* kh = kg + (size_t)bh * SS * HD;
    const unsigned short* vh = vg + (size_t)bh * SS * HD;

    __shared__ float          sc[16][WKEYS];
    __shared__ unsigned short pr[16][WPAD];

    const float NEGBIG = -3.0e38f;
    for (int t = lane; t < 16 * WKEYS; t += 32)
        sc[t / WKEYS][t % WKEYS] = NEGBIG;
    __syncthreads();

    const v16bf aq0 = load_a_frag(qh + (size_t)q0 * HD, HD, lane);
    const v16bf aq1 = load_a_frag(qh + (size_t)q0 * HD + 32, HD, lane);

    for (int kt = 0; kt < 17; ++kt) {
        const int ks0 = q0 - BLK + kt * 16;
        if (ks0 + 15 < 0 || ks0 >= SS) continue;

        const int col  = lane & 15;
        int key  = ks0 + col;
        int keyc = key < 0 ? 0 : (key >= SS ? SS - 1 : key);
        const unsigned short* kp = kh + (size_t)keyc * HD + ((lane >> 4) << 4);
        v8u u0, u1;
#pragma unroll
        for (int v = 0; v < 8; ++v) {
            u0[v] = *(const unsigned int*)(kp + (v << 1));
            u1[v] = *(const unsigned int*)(kp + 32 + (v << 1));
        }
        FragCast c0; c0.u = u0;
        FragCast c1; c1.u = u1;

        v8f s = (v8f)0.f;
        s = wmma_bf16(aq0, c0.b, s);
        s = wmma_bf16(aq1, c1.b, s);

        const int mbase = (lane >> 4) << 3;
#pragma unroll
        for (int r = 0; r < 8; ++r) {
            int m  = mbase + r;
            int qi = q0 + m;
            int kj = ks0 + col;
            bool ok = (kj >= 0) && (kj < SS) && (abs(qi - kj) <= BLK);
            sc[m][kt * 16 + col] = ok ? s[r] * 0.125f : NEGBIG;
        }
    }
    __syncthreads();

    if (lane < 16) {
        const int row = lane;
        const int qi  = q0 + row;
        float mx = NEGBIG;
        for (int j = 0; j < WKEYS; ++j) mx = fmaxf(mx, sc[row][j]);
        float sum = 0.f;
        for (int j = 0; j < WKEYS; ++j) {
            float e = __expf(sc[row][j] - mx);
            sc[row][j] = e;
            sum += e;
        }
        const float inv = 1.0f / sum;
        const float wsc = inv * (1.0f / (float)HH);
        for (int j = 0; j < WKEYS; ++j) {
            float p = sc[row][j] * inv;
            pr[row][j] = f2bf(p);
            int kj = q0 - BLK + j;
            if (kj >= 0 && kj < SS)
                atomicAdd(&attnW[((size_t)b_ * SS + qi) * SS + kj], sc[row][j] * wsc);
        }
        for (int j = WKEYS; j < WPAD; ++j) pr[row][j] = 0;
    }
    __syncthreads();

    v8f acc[4];
#pragma unroll
    for (int j = 0; j < 4; ++j) acc[j] = (v8f)0.f;

    for (int kc = 0; kc < WPAD / 32; ++kc) {
        v16bf a = load_a_frag(&pr[0][kc * 32], WPAD, lane);
        const int kb = kc * 32 + ((lane >> 4) << 4);
#pragma unroll
        for (int j = 0; j < 4; ++j) {
            const int hd = j * 16 + (lane & 15);
            v8u u;
#pragma unroll
            for (int v = 0; v < 8; ++v) {
                int kidx = kb + (v << 1);
                int k0 = q0 - BLK + kidx;
                int k1 = k0 + 1;
                int k0c = k0 < 0 ? 0 : (k0 >= SS ? SS - 1 : k0);
                int k1c = k1 < 0 ? 0 : (k1 >= SS ? SS - 1 : k1);
                unsigned int lo = vh[(size_t)k0c * HD + hd];
                unsigned int hi = vh[(size_t)k1c * HD + hd];
                u[v] = lo | (hi << 16);
            }
            FragCast cb; cb.u = u;
            acc[j] = wmma_bf16(a, cb.b, acc[j]);
        }
    }

    const int mbase = (lane >> 4) << 3;
#pragma unroll
    for (int j = 0; j < 4; ++j) {
        const int hd = j * 16 + (lane & 15);
#pragma unroll
        for (int r = 0; r < 8; ++r) {
            int qi = q0 + mbase + r;
            ctx[((size_t)b_ * SS + qi) * DD + h_ * HD + hd] = f2bf(acc[j][r]);
        }
    }
}

// ---------------------------------------------------------------------------
// Host-side launcher
// ---------------------------------------------------------------------------
extern "C" void kernel_launch(void* const* d_in, const int* in_sizes, int n_in,
                              void* d_out, int out_size, void* d_ws, size_t ws_size,
                              hipStream_t stream) {
    const float* x    = (const float*)d_in[0];
    const float* ipw  = (const float*)d_in[1];   // [3D, D]
    const float* ipb  = (const float*)d_in[2];
    const float* opw  = (const float*)d_in[3];   // [D, D]
    const float* opb  = (const float*)d_in[4];
    const float* ln1g = (const float*)d_in[5];
    const float* ln1b = (const float*)d_in[6];
    const float* ln2g = (const float*)d_in[7];
    const float* ln2b = (const float*)d_in[8];
    const float* w1   = (const float*)d_in[9];   // [FF, D]
    const float* b1   = (const float*)d_in[10];
    const float* w2   = (const float*)d_in[11];  // [D, FF]
    const float* b2   = (const float*)d_in[12];

    float* out_x    = (float*)d_out;
    float* out_attn = out_x + (size_t)BB * SS * DD;

    char* w = (char*)d_ws;
    unsigned short* xn    = (unsigned short*)(w);                          // 8 MB
    unsigned short* qb    = (unsigned short*)(w + ((size_t)8   << 20));    // 8 MB
    unsigned short* kb    = (unsigned short*)(w + ((size_t)16  << 20));    // 8 MB
    unsigned short* vb    = (unsigned short*)(w + ((size_t)24  << 20));    // 8 MB
    unsigned short* ctx   = (unsigned short*)(w + ((size_t)32  << 20));    // 8 MB
    float*          x1    = (float*)         (w + ((size_t)40  << 20));    // 16 MB
    unsigned short* xn2   = (unsigned short*)(w + ((size_t)56  << 20));    // 8 MB
    unsigned short* hbuf  = (unsigned short*)(w + ((size_t)64  << 20));    // 32 MB
    unsigned short* ipwb  = (unsigned short*)(w + ((size_t)96  << 20));    // 6 MB
    unsigned short* opwb  = (unsigned short*)(w + ((size_t)102 << 20));    // 2 MB
    unsigned short* w1b   = (unsigned short*)(w + ((size_t)104 << 20));    // 8 MB
    unsigned short* w2b   = (unsigned short*)(w + ((size_t)112 << 20));    // 8 MB

    // 0) weights -> bf16
    cast_bf16_kernel<<<(3 * DD * DD + 255) / 256, 256, 0, stream>>>(ipw, ipwb, 3 * DD * DD);
    cast_bf16_kernel<<<(DD * DD + 255) / 256, 256, 0, stream>>>(opw, opwb, DD * DD);
    cast_bf16_kernel<<<(FF_ * DD + 255) / 256, 256, 0, stream>>>(w1, w1b, FF_ * DD);
    cast_bf16_kernel<<<(DD * FF_ + 255) / 256, 256, 0, stream>>>(w2, w2b, DD * FF_);

    // 1) LN1 -> bf16
    ln_bf16_kernel<<<NTOK, 256, 0, stream>>>(x, ln1g, ln1b, xn, DD);

    // 2) QKV projection with scatter epilogue
    {
        GemmArgs ga{};
        ga.A = xn; ga.W = ipwb; ga.bias = ipb;
        ga.q = qb; ga.k = kb; ga.v = vb;
        ga.M = NTOK; ga.N = 3 * DD; ga.K = DD;
        gemm_wmma_kernel<0><<<dim3((3 * DD) / 64, NTOK / 128), 256, 0, stream>>>(ga);
    }

    // 3) zero attn-weights output
    {
        int n = BB * SS * SS;
        zero_f32_kernel<<<(n + 255) / 256, 256, 0, stream>>>(out_attn, n);
    }

    // 4) banded attention
    attn_wmma_kernel<<<dim3(SS / 16, BB * HH), 32, 0, stream>>>(qb, kb, vb, ctx, out_attn);

    // 5) out-proj + residual -> x1
    {
        GemmArgs ga{};
        ga.A = ctx; ga.W = opwb; ga.bias = opb;
        ga.resid = x; ga.outf = x1;
        ga.M = NTOK; ga.N = DD; ga.K = DD;
        gemm_wmma_kernel<2><<<dim3(DD / 64, NTOK / 128), 256, 0, stream>>>(ga);
    }

    // 6) LN2 -> bf16
    ln_bf16_kernel<<<NTOK, 256, 0, stream>>>(x1, ln2g, ln2b, xn2, DD);

    // 7) FFN1 with exact GELU -> bf16
    {
        GemmArgs ga{};
        ga.A = xn2; ga.W = w1b; ga.bias = b1; ga.outb = hbuf;
        ga.M = NTOK; ga.N = FF_; ga.K = DD;
        gemm_wmma_kernel<1><<<dim3(FF_ / 64, NTOK / 128), 256, 0, stream>>>(ga);
    }

    // 8) FFN2 + residual -> d_out x
    {
        GemmArgs ga{};
        ga.A = hbuf; ga.W = w2b; ga.bias = b2;
        ga.resid = x1; ga.outf = out_x;
        ga.M = NTOK; ga.N = DD; ga.K = FF_;
        gemm_wmma_kernel<2><<<dim3(DD / 64, NTOK / 128), 256, 0, stream>>>(ga);
    }
}
